// TransLayer_17746804867625
// MI455X (gfx1250) — compile-verified, matrix operand order checked by reading, not run on hardware
//
#include <hip/hip_runtime.h>
#include <hip/hip_bf16.h>

// ---------------------------------------------------------------------------
// Nystrom-attention transformer layer for MI455X (gfx1250, wave32, WMMA).
// Heavy GEMMs: v_wmma_f32_16x16x32_bf16, double-buffered LDS tiles fed by
// GLOBAL_LOAD_ASYNC_TO_LDS_B128 (ASYNCcnt) when operands are bf16 in memory.
// B=4, N=4096, D=512, H=8, DH=64, M=256, KERNEL=33, ITERS=6.
// ---------------------------------------------------------------------------

#define Bn   4
#define Nn   4096
#define Dn   512
#define Hn   8
#define DHn  64
#define Mn   256
#define KW   33
#define NITER 6
#define BHn  (Bn*Hn)       // 32
#define BNn  (Bn*Nn)       // 16384

typedef __attribute__((ext_vector_type(16))) __bf16 v16bf;
typedef __attribute__((ext_vector_type(8)))  float  v8f;
typedef __attribute__((ext_vector_type(4)))  int    v4i;

struct __align__(16) bf16x8 { __bf16 v[8]; };

#if defined(__gfx1250__) && __has_builtin(__builtin_amdgcn_global_load_async_to_lds_b128)
#define ASYNC_OK 1
#else
#define ASYNC_OK 0
#endif

__device__ __forceinline__ void wait_async() {
#if defined(__gfx1250__)
#if __has_builtin(__builtin_amdgcn_s_wait_asynccnt)
  __builtin_amdgcn_s_wait_asynccnt(0);
#else
  asm volatile("s_wait_asynccnt 0x0" ::: "memory");
#endif
#endif
}

__device__ __forceinline__ void async_cp16(const void* g, void* l) {
#if ASYNC_OK
  // builtin signature (probe-derived): (as1 int4*, as3 int4*, imm offset, cpol)
  __builtin_amdgcn_global_load_async_to_lds_b128(
      (__attribute__((address_space(1))) v4i*)g,
      (__attribute__((address_space(3))) v4i*)l, 0, 0);
#else
  *(bf16x8*)l = *(const bf16x8*)g;
#endif
}

// ---------------------------------------------------------------------------
// LayerNorm: one block per row of 512, 2 elems/thread, write bf16.
// ---------------------------------------------------------------------------
__global__ __launch_bounds__(256)
void k_ln(const float* __restrict__ x, const float* __restrict__ g,
          const float* __restrict__ be, __bf16* __restrict__ xn) {
  int row = blockIdx.x, tid = threadIdx.x;
  const float* xr = x + (long)row * Dn;
  float a = xr[tid], b = xr[tid + 256];
  float s = a + b, ss = a * a + b * b;
  #pragma unroll
  for (int m = 16; m; m >>= 1) { s += __shfl_xor(s, m); ss += __shfl_xor(ss, m); }
  __shared__ float r1[8], r2[8];
  __shared__ float mu_s, rs_s;
  if ((tid & 31) == 0) { r1[tid >> 5] = s; r2[tid >> 5] = ss; }
  __syncthreads();
  if (tid == 0) {
    float S = 0.f, SS = 0.f;
    for (int i = 0; i < 8; ++i) { S += r1[i]; SS += r2[i]; }
    float mu = S / (float)Dn;
    float var = SS / (float)Dn - mu * mu;
    mu_s = mu; rs_s = rsqrtf(var + 1e-5f);
  }
  __syncthreads();
  float mu = mu_s, rs = rs_s;
  __bf16* o = xn + (long)row * Dn;
  o[tid]       = (__bf16)((a - mu) * rs * g[tid]       + be[tid]);
  o[tid + 256] = (__bf16)((b - mu) * rs * g[tid + 256] + be[tid + 256]);
}

// ---------------------------------------------------------------------------
// Generic batched WMMA GEMM:  C[M_,N_] = alpha * A[M_,K_] x B  (+bias +resid)
//   A row-major (TA), B row-major [K_,N_] or (BT) stored as [N_,K_].
//   Double-buffered 64x64 tile; bf16 operands staged via async b128 copies
//   (f32 operands converted through VGPRs). 8 waves, each 16x32 out.
// ---------------------------------------------------------------------------
template <typename TA, typename TB, bool BT>
__global__ __launch_bounds__(256)
void k_gemm(const void* __restrict__ Av, const void* __restrict__ Bv,
            float* __restrict__ C, int M_, int N_, int K_,
            long sA, long sB, long sC, float alpha,
            const float* __restrict__ bias, const float* __restrict__ resid) {
  const TA* A = (const TA*)Av + (long)blockIdx.z * sA;
  const TB* B = (const TB*)Bv + (long)blockIdx.z * sB;
  float*    Cb = C + (long)blockIdx.z * sC;
  const float* Rb = resid ? resid + (long)blockIdx.z * sC : nullptr;

  int tid = threadIdx.x, wid = tid >> 5, lane = tid & 31;
  int row0 = blockIdx.y * 64, col0 = blockIdx.x * 64;
  int wm = wid & 3, wn = wid >> 2;                 // 4x2 wave grid
  int half = lane >> 4, l16 = lane & 15;

  __shared__ __align__(16) __bf16 As[2][64][32];   // [buf][row][k]
  __shared__ __align__(16) __bf16 Bs[2][64][32];   // [buf][col][k] (B^T tile)

  // --- tile staging: one 16B chunk per thread per tile ---
  auto stageA = [&](int k0, int buf) {
    int r = tid >> 2, part = (tid & 3) * 8;        // 64 rows x 4 chunks
    const TA* src = A + (long)(row0 + r) * K_ + (k0 + part);
    __bf16* dst = &As[buf][r][part];
    if constexpr (sizeof(TA) == 2) {
      async_cp16(src, dst);
    } else {
      float4 f0 = *(const float4*)(const void*)src;
      float4 f1 = *(const float4*)(const void*)(src + 4);
      bf16x8 t;
      t.v[0]=(__bf16)f0.x; t.v[1]=(__bf16)f0.y; t.v[2]=(__bf16)f0.z; t.v[3]=(__bf16)f0.w;
      t.v[4]=(__bf16)f1.x; t.v[5]=(__bf16)f1.y; t.v[6]=(__bf16)f1.z; t.v[7]=(__bf16)f1.w;
      *(bf16x8*)dst = t;
    }
  };
  auto stageB = [&](int k0, int buf) {
    if constexpr (BT) {                            // B stored [N_,K_]: k-contig
      int c = tid >> 2, part = (tid & 3) * 8;
      const TB* src = B + (long)(col0 + c) * K_ + (k0 + part);
      __bf16* dst = &Bs[buf][c][part];
      if constexpr (sizeof(TB) == 2) {
        async_cp16(src, dst);
      } else {
        float4 f0 = *(const float4*)(const void*)src;
        float4 f1 = *(const float4*)(const void*)(src + 4);
        bf16x8 t;
        t.v[0]=(__bf16)f0.x; t.v[1]=(__bf16)f0.y; t.v[2]=(__bf16)f0.z; t.v[3]=(__bf16)f0.w;
        t.v[4]=(__bf16)f1.x; t.v[5]=(__bf16)f1.y; t.v[6]=(__bf16)f1.z; t.v[7]=(__bf16)f1.w;
        *(bf16x8*)dst = t;
      }
    } else {                                       // B stored [K_,N_]: n-contig
      int kk = tid >> 3, cg = (tid & 7) * 8;       // 32 k x 8 col-groups
      const TB* src = B + (long)(k0 + kk) * N_ + (col0 + cg);
      if constexpr (sizeof(TB) == 2) {
        bf16x8 t = *(const bf16x8*)(const void*)src;
        #pragma unroll
        for (int j = 0; j < 8; ++j) Bs[buf][cg + j][kk] = t.v[j];
      } else {
        float4 f0 = *(const float4*)(const void*)src;
        float4 f1 = *(const float4*)(const void*)(src + 4);
        float f[8] = {f0.x, f0.y, f0.z, f0.w, f1.x, f1.y, f1.z, f1.w};
        #pragma unroll
        for (int j = 0; j < 8; ++j) Bs[buf][cg + j][kk] = (__bf16)f[j];
      }
    }
  };

  v8f acc0 = {}, acc1 = {};
  int nk = K_ >> 5;

  stageA(0, 0); stageB(0, 0);
  wait_async();
  __syncthreads();

  for (int ks = 0; ks < nk; ++ks) {
    int cur = ks & 1;
    if (ks + 1 < nk) { stageA((ks + 1) << 5, cur ^ 1); stageB((ks + 1) << 5, cur ^ 1); }

    v16bf a, b0, b1;
    int ar = 16 * wm + l16;
    int bc = 32 * wn + l16;
    #pragma unroll
    for (int j = 0; j < 8; ++j) {
      a[j]      = As[cur][ar][half * 8 + j];
      a[8 + j]  = As[cur][ar][16 + half * 8 + j];
      b0[j]     = Bs[cur][bc][half * 8 + j];
      b0[8 + j] = Bs[cur][bc][16 + half * 8 + j];
      b1[j]     = Bs[cur][bc + 16][half * 8 + j];
      b1[8 + j] = Bs[cur][bc + 16][16 + half * 8 + j];
    }
    acc0 = __builtin_amdgcn_wmma_f32_16x16x32_bf16(false, a, false, b0,
                                                   (short)0, acc0, false, false);
    acc1 = __builtin_amdgcn_wmma_f32_16x16x32_bf16(false, a, false, b1,
                                                   (short)0, acc1, false, false);
    wait_async();
    __syncthreads();
  }

  // C layout: VGPR i -> row = base + i + 8*half ; col = base + (lane&15)
  int crow = row0 + 16 * wm + 8 * half;
  int ccol = col0 + 32 * wn + l16;
  #pragma unroll
  for (int i = 0; i < 8; ++i) {
    long i0 = (long)(crow + i) * N_ + ccol;
    long i1 = i0 + 16;
    float v0 = alpha * acc0[i];
    float v1 = alpha * acc1[i];
    if (bias)  { v0 += bias[ccol];  v1 += bias[ccol + 16]; }
    if (Rb)    { v0 += Rb[i0];      v1 += Rb[i1]; }
    Cb[i0] = v0;
    Cb[i1] = v1;
  }
}

// ---------------------------------------------------------------------------
// Split fused QKV (f32 [BN,1536]) into q(scaled)/k/v bf16 [B*H, N, DH].
// ---------------------------------------------------------------------------
__global__ __launch_bounds__(256)
void k_split(const float* __restrict__ qkv, __bf16* __restrict__ q,
             __bf16* __restrict__ k, __bf16* __restrict__ v) {
  int i = blockIdx.x * 256 + threadIdx.x;          // over BN*D
  if (i >= BNn * Dn) return;
  int d = i & (Dn - 1); int bn = i >> 9;
  int b = bn >> 12;     int n = bn & (Nn - 1);
  int h = d >> 6;       int dh = d & 63;
  long src = (long)bn * (3 * Dn) + d;
  long dst = ((long)(b * Hn + h) << 18) + ((long)n << 6) + dh;
  q[dst] = (__bf16)(qkv[src] * 0.125f);            // DH^-0.5
  k[dst] = (__bf16)(qkv[src + Dn]);
  v[dst] = (__bf16)(qkv[src + 2 * Dn]);
}

// Landmark pooling: dst[bh][m][dh] = mean_{j<16} src[bh][16m+j][dh]
__global__ __launch_bounds__(256)
void k_pool(const __bf16* __restrict__ src, __bf16* __restrict__ dst) {
  int i = blockIdx.x * 256 + threadIdx.x;          // 32*256*64
  if (i >= BHn * Mn * DHn) return;
  int dh = i & 63; int m = (i >> 6) & (Mn - 1); int bh = i >> 14;
  long base = ((long)bh << 18) + ((long)(m * 16) << 6) + dh;
  float s = 0.f;
  #pragma unroll
  for (int j = 0; j < 16; ++j) s += (float)src[base + ((long)j << 6)];
  dst[i] = (__bf16)(s * 0.0625f);
}

// ---------------------------------------------------------------------------
// Row softmax (f32 in -> bf16 out). One 256-thread block per row; row cached
// in registers (L/256 <= 16). wave32 shuffle + LDS reductions.
// ---------------------------------------------------------------------------
__global__ __launch_bounds__(256)
void k_softmax(const float* __restrict__ in, __bf16* __restrict__ out, int L) {
  long row = blockIdx.x;
  const float* x = in + row * (long)L;
  __bf16* o = out + row * (long)L;
  int tid = threadIdx.x, cnt = L >> 8;
  float r[16];
  float mx = -3.4e38f;
  for (int i = 0; i < cnt; ++i) { r[i] = x[tid + (i << 8)]; mx = fmaxf(mx, r[i]); }
  #pragma unroll
  for (int m = 16; m; m >>= 1) mx = fmaxf(mx, __shfl_xor(mx, m));
  __shared__ float sh[8];
  if ((tid & 31) == 0) sh[tid >> 5] = mx;
  __syncthreads();
  float bm = sh[0];
  #pragma unroll
  for (int i = 1; i < 8; ++i) bm = fmaxf(bm, sh[i]);
  __syncthreads();
  float sum = 0.f;
  for (int i = 0; i < cnt; ++i) { r[i] = __expf(r[i] - bm); sum += r[i]; }
  #pragma unroll
  for (int m = 16; m; m >>= 1) sum += __shfl_xor(sum, m);
  if ((tid & 31) == 0) sh[tid >> 5] = sum;
  __syncthreads();
  float bs = 0.f;
  #pragma unroll
  for (int i = 0; i < 8; ++i) bs += sh[i];
  float inv = 1.f / bs;
  for (int i = 0; i < cnt; ++i) o[tid + (i << 8)] = (__bf16)(r[i] * inv);
}

// ---------------------------------------------------------------------------
// Moore-Penrose pinv helpers. attn2 rows are softmax output (>=0), so
// positive-float atomicMax via int bit pattern is order-preserving.
// ---------------------------------------------------------------------------
__global__ void k_zero2(float* s) { if (threadIdx.x < 2) s[threadIdx.x] = 0.f; }

__global__ __launch_bounds__(256)
void k_rowabsmax(const __bf16* __restrict__ a, float* __restrict__ s) {
  int i = blockIdx.x * 256 + threadIdx.x;          // batch*row = 32*256
  if (i >= BHn * Mn) return;
  const __bf16* row = a + (long)i * Mn;
  float sum = 0.f;
  for (int j = 0; j < Mn; ++j) sum += (float)row[j];
  atomicMax((int*)&s[0], __float_as_int(sum));
}

__global__ __launch_bounds__(256)
void k_colabsmax(const __bf16* __restrict__ a, float* __restrict__ s) {
  int i = blockIdx.x * 256 + threadIdx.x;          // batch*col
  if (i >= BHn * Mn) return;
  int b = i >> 8, c = i & (Mn - 1);
  const __bf16* p = a + (long)b * Mn * Mn + c;
  float sum = 0.f;
  for (int j = 0; j < Mn; ++j) sum += (float)p[(long)j * Mn];
  atomicMax((int*)&s[1], __float_as_int(sum));
}

// z0 = a^T / (rowmax*colmax)
__global__ __launch_bounds__(256)
void k_tr_scale(const __bf16* __restrict__ a, const float* __restrict__ s,
                float* __restrict__ z) {
  int i = blockIdx.x * 256 + threadIdx.x;          // 32*256*256
  if (i >= BHn * Mn * Mn) return;
  int ij = i & (Mn * Mn - 1); int b = i >> 16;
  int r = ij >> 8, c = ij & (Mn - 1);
  float inv = 1.f / (s[0] * s[1]);
  z[i] = (float)a[(long)b * Mn * Mn + (long)c * Mn + r] * inv;
}

// out = alphaI*I - in   (batched MxM)
__global__ __launch_bounds__(256)
void k_ialpha(const float* __restrict__ in, float* __restrict__ out,
              float aI, int total) {
  int i = blockIdx.x * 256 + threadIdx.x;
  if (i >= total) return;
  int ij = i & (Mn * Mn - 1);
  int r = ij >> 8, c = ij & (Mn - 1);
  out[i] = (r == c ? aI : 0.f) - in[i];
}

// ---------------------------------------------------------------------------
// Depthwise residual conv over sequence dim (kernel 33, per head), add into
// the attention output [B*H, N, DH].
// ---------------------------------------------------------------------------
__global__ __launch_bounds__(256)
void k_conv_add(const __bf16* __restrict__ v, const float* __restrict__ w,
                float* __restrict__ oh) {
  int i = blockIdx.x * 256 + threadIdx.x;          // 32*4096*64
  if (i >= BHn * Nn * DHn) return;
  int dh = i & 63; int n = (i >> 6) & (Nn - 1); int bh = i >> 18;
  int h = bh & (Hn - 1);
  float s = 0.f;
  #pragma unroll
  for (int t = 0; t < KW; ++t) {
    int nn = n + t - (KW / 2);
    if (nn >= 0 && nn < Nn)
      s += w[h * KW + t] * (float)v[((long)bh << 18) + ((long)nn << 6) + dh];
  }
  oh[i] += s;
}

// Merge heads: [B*H, N, DH] -> [B*N, D]
__global__ __launch_bounds__(256)
void k_merge(const float* __restrict__ oh, float* __restrict__ m) {
  int i = blockIdx.x * 256 + threadIdx.x;
  if (i >= BHn * Nn * DHn) return;
  int dh = i & 63; int n = (i >> 6) & (Nn - 1); int bh = i >> 18;
  int b = bh >> 3, h = bh & (Hn - 1);
  m[((long)(b * Nn + n)) * Dn + h * DHn + dh] = oh[i];
}

// ---------------------------------------------------------------------------
// Host orchestration
// ---------------------------------------------------------------------------
extern "C" void kernel_launch(void* const* d_in, const int* in_sizes, int n_in,
                              void* d_out, int out_size, void* d_ws, size_t ws_size,
                              hipStream_t stream) {
  const float* x     = (const float*)d_in[0];   // [B,N,D]
  const float* gamma = (const float*)d_in[1];
  const float* beta  = (const float*)d_in[2];
  const float* w_qkv = (const float*)d_in[3];   // [D,3D]
  const float* res_w = (const float*)d_in[4];   // [H,1,33,1]
  const float* w_out = (const float*)d_in[5];   // [D,D]
  const float* b_out = (const float*)d_in[6];   // [D]
  float* out = (float*)d_out;

  char* p = (char*)d_ws;
  auto alloc = [&](size_t bytes) -> void* {
    void* r = (void*)p; p += (bytes + 255) & ~(size_t)255; return r;
  };

  // BIG0 is reused sequentially: qkv f32 -> logits1 -> logits3 -> out1
  float*  BIG0     = (float*) alloc((size_t)BHn * Nn * Mn * 4);   // 134 MB
  __bf16* attn1_bf = (__bf16*)alloc((size_t)BHn * Nn * Mn * 2);   //  67 MB
  __bf16* attn3_bf = (__bf16*)alloc((size_t)BHn * Mn * Nn * 2);   //  67 MB
  __bf16* xn_bf    = (__bf16*)alloc((size_t)BNn * Dn * 2);
  __bf16* q_bf     = (__bf16*)alloc((size_t)BHn * Nn * DHn * 2);
  __bf16* k_bf     = (__bf16*)alloc((size_t)BHn * Nn * DHn * 2);
  __bf16* v_bf     = (__bf16*)alloc((size_t)BHn * Nn * DHn * 2);
  __bf16* ql_bf    = (__bf16*)alloc((size_t)BHn * Mn * DHn * 2);
  __bf16* kl_bf    = (__bf16*)alloc((size_t)BHn * Mn * DHn * 2);
  float*  log2_f   = (float*) alloc((size_t)BHn * Mn * Mn * 4);
  __bf16* attn2_bf = (__bf16*)alloc((size_t)BHn * Mn * Mn * 2);
  float*  z0       = (float*) alloc((size_t)BHn * Mn * Mn * 4);
  float*  z1       = (float*) alloc((size_t)BHn * Mn * Mn * 4);
  float*  az       = (float*) alloc((size_t)BHn * Mn * Mn * 4);
  float*  tbuf     = (float*) alloc((size_t)BHn * Mn * Mn * 4);
  float*  ubuf     = (float*) alloc((size_t)BHn * Mn * Mn * 4);
  float*  kv       = (float*) alloc((size_t)BHn * Mn * DHn * 4);
  float*  oh       = (float*) alloc((size_t)BHn * Nn * DHn * 4);
  float*  merged   = (float*) alloc((size_t)BNn * Dn * 4);
  float*  scal     = (float*) alloc(256);

  dim3 blk(256);

  // 1) LayerNorm -> xn (bf16)
  k_ln<<<BNn, blk, 0, stream>>>(x, gamma, beta, xn_bf);

  // 2) QKV GEMM: xn[16384,512] x w_qkv[512,1536] -> BIG0 f32
  k_gemm<__bf16, float, false><<<dim3(24, 256, 1), blk, 0, stream>>>(
      xn_bf, w_qkv, BIG0, BNn, 3 * Dn, Dn, 0, 0, 0, 1.f, nullptr, nullptr);

  // 3) split heads (q scaled) + landmark pooling
  k_split<<<(BNn * Dn) / 256, blk, 0, stream>>>(BIG0, q_bf, k_bf, v_bf);
  k_pool<<<(BHn * Mn * DHn) / 256, blk, 0, stream>>>(q_bf, ql_bf);
  k_pool<<<(BHn * Mn * DHn) / 256, blk, 0, stream>>>(k_bf, kl_bf);

  // 4) attn1 = softmax(q @ kl^T)  [32, 4096, 256]
  k_gemm<__bf16, __bf16, true><<<dim3(4, 64, BHn), blk, 0, stream>>>(
      q_bf, kl_bf, BIG0, Nn, Mn, DHn,
      (long)Nn * DHn, (long)Mn * DHn, (long)Nn * Mn, 1.f, nullptr, nullptr);
  k_softmax<<<BHn * Nn, blk, 0, stream>>>(BIG0, attn1_bf, Mn);

  // 5) attn2 = softmax(ql @ kl^T) [32, 256, 256]
  k_gemm<__bf16, __bf16, true><<<dim3(4, 4, BHn), blk, 0, stream>>>(
      ql_bf, kl_bf, log2_f, Mn, Mn, DHn,
      (long)Mn * DHn, (long)Mn * DHn, (long)Mn * Mn, 1.f, nullptr, nullptr);
  k_softmax<<<BHn * Mn, blk, 0, stream>>>(log2_f, attn2_bf, Mn);

  // 6) attn3 = softmax(ql @ k^T) [32, 256, 4096]
  k_gemm<__bf16, __bf16, true><<<dim3(64, 4, BHn), blk, 0, stream>>>(
      ql_bf, k_bf, BIG0, Mn, Nn, DHn,
      (long)Mn * DHn, (long)Nn * DHn, (long)Mn * Nn, 1.f, nullptr, nullptr);
  k_softmax<<<BHn * Mn, blk, 0, stream>>>(BIG0, attn3_bf, Nn);

  // 7) Newton-Schulz pseudo-inverse of attn2
  k_zero2<<<1, 32, 0, stream>>>(scal);
  k_rowabsmax<<<(BHn * Mn) / 256, blk, 0, stream>>>(attn2_bf, scal);
  k_colabsmax<<<(BHn * Mn) / 256, blk, 0, stream>>>(attn2_bf, scal);
  k_tr_scale<<<(BHn * Mn * Mn) / 256, blk, 0, stream>>>(attn2_bf, scal, z0);

  float* zp = z0; float* zq = z1;
  const long sMM = (long)Mn * Mn;
  const int totMM = BHn * Mn * Mn;
  for (int it = 0; it < NITER; ++it) {
    k_gemm<__bf16, float, false><<<dim3(4, 4, BHn), blk, 0, stream>>>(
        attn2_bf, zp, az, Mn, Mn, Mn, sMM, sMM, sMM, 1.f, nullptr, nullptr);
    k_ialpha<<<totMM / 256, blk, 0, stream>>>(az, tbuf, 7.f, totMM);
    k_gemm<float, float, false><<<dim3(4, 4, BHn), blk, 0, stream>>>(
        az, tbuf, ubuf, Mn, Mn, Mn, sMM, sMM, sMM, 1.f, nullptr, nullptr);
    k_ialpha<<<totMM / 256, blk, 0, stream>>>(ubuf, tbuf, 15.f, totMM);
    k_gemm<float, float, false><<<dim3(4, 4, BHn), blk, 0, stream>>>(
        az, tbuf, ubuf, Mn, Mn, Mn, sMM, sMM, sMM, 1.f, nullptr, nullptr);
    k_ialpha<<<totMM / 256, blk, 0, stream>>>(ubuf, tbuf, 13.f, totMM);
    k_gemm<float, float, false><<<dim3(4, 4, BHn), blk, 0, stream>>>(
        zp, tbuf, zq, Mn, Mn, Mn, sMM, sMM, sMM, 0.25f, nullptr, nullptr);
    float* t = zp; zp = zq; zq = t;
  }

  // 8) kv = attn3 @ v  [32, 256, 64]
  k_gemm<__bf16, __bf16, false><<<dim3(1, 4, BHn), blk, 0, stream>>>(
      attn3_bf, v_bf, kv, Mn, DHn, Nn,
      (long)Mn * Nn, (long)Nn * DHn, (long)Mn * DHn, 1.f, nullptr, nullptr);

  // 9) out1 = attn1 @ z  [32, 4096, 256] -> BIG0
  k_gemm<__bf16, float, false><<<dim3(4, 64, BHn), blk, 0, stream>>>(
      attn1_bf, zp, BIG0, Nn, Mn, Mn,
      (long)Nn * Mn, sMM, (long)Nn * Mn, 1.f, nullptr, nullptr);

  // 10) out_heads = out1 @ kv  [32, 4096, 64]
  k_gemm<float, float, false><<<dim3(1, 64, BHn), blk, 0, stream>>>(
      BIG0, kv, oh, Nn, DHn, Mn,
      (long)Nn * Mn, (long)Mn * DHn, (long)Nn * DHn, 1.f, nullptr, nullptr);

  // 11) depthwise conv residual (+=), merge heads
  k_conv_add<<<(BHn * Nn * DHn) / 256, blk, 0, stream>>>(v_bf, res_w, oh);
  k_merge<<<(BHn * Nn * DHn) / 256, blk, 0, stream>>>(oh, merged);

  // 12) output projection + bias + input residual -> d_out
  k_gemm<float, float, false><<<dim3(8, 256, 1), blk, 0, stream>>>(
      merged, w_out, out, BNn, Dn, Dn, 0, 0, 0, 1.f, b_out, x);
}
